// LSTMAutoencoder_65481071411026
// MI455X (gfx1250) — matrix operand order repaired and checked by reading, not compiled, not run on hardware
//
#include <hip/hip_runtime.h>
#include <hip/hip_bf16.h>
#include <stdint.h>

// ---------------- problem dims ----------------
#define BB 256
#define SS 512
#define FF 64
#define HH 256
#define EE 128
// d_out layout: recon [B,S,F] | emb [B,E] | weights [B,S]
#define RECON_OFF 0
#define EMB_OFF   (BB * SS * FF)                 // 8388608
#define WGT_OFF   (EMB_OFF + BB * EE)            // 8421376

// packed-weight dword offsets inside d_ws
#define PK_ENC_WIH 0
#define PK_ENC_WHH 32768
#define PK_DEC_WIH 163840
#define PK_DEC_WHH 196608
#define PK_FC      327680
#define PK_DH      344064
#define PK_DC      360448
#define PK_OUT     376832

typedef __attribute__((ext_vector_type(16))) __bf16 v16bf;
typedef __attribute__((ext_vector_type(8)))  float  v8f;

union FragU { uint32_t u[8]; uint4 q[2]; v16bf v; };

// ---- WMMA wrapper: D = A(16x32 bf16) * B(32x16 bf16) + C(16x16 f32) ----
__device__ __forceinline__ v8f wmma_bf16(v16bf a, v16bf b, v8f c) {
    return __builtin_amdgcn_wmma_f32_16x16x32_bf16(false, a, false, b, (short)0, c,
                                                   false, false);
}

// B fragment, pre-swizzled in global memory: frag = 256 dwords, lane gets 8.
__device__ __forceinline__ v16bf load_b_frag(const uint32_t* __restrict__ pk,
                                             int fragIdx, int lane) {
    FragU f;
    const uint4* p = (const uint4*)(pk + (size_t)fragIdx * 256 + (size_t)lane * 8);
    f.q[0] = p[0];
    f.q[1] = p[1];
    return f.v;
}

// A fragment from LDS (row-major bf16, strideDw dwords per row).
// ISA 16-bit A 16x32 layout: lane l holds row M=l%16; vgpr v holds K pair
// K0 = kt*32 + (v/4)*16 + (l/16)*8 + (v%4)*2  -> two contiguous 16B groups.
__device__ __forceinline__ v16bf load_a_frag(const uint32_t* sdw, int strideDw,
                                             int kt, int lane) {
    FragU f;
    const int M = lane & 15;
    const int half = lane >> 4;
    const uint32_t* p = sdw + M * strideDw + kt * 16 + half * 4;
    f.q[0] = *(const uint4*)(p);      // K group 0 (dwords +0..3)
    f.q[1] = *(const uint4*)(p + 8);  // K group 1 (dwords +8..11)
    return f.v;
}

__device__ __forceinline__ float fsig(float x) {
    return __builtin_amdgcn_rcpf(1.0f + __expf(-x));
}
__device__ __forceinline__ float ftanh_(float x) {
    const float e = __expf(-2.0f * x);
    return (1.0f - e) * __builtin_amdgcn_rcpf(1.0f + e);
}

__device__ __forceinline__ uint32_t pk2bf(float a, float b) {
    __hip_bfloat16 ha = __float2bfloat16(a);
    __hip_bfloat16 hb = __float2bfloat16(b);
    uint16_t ua, ub;
    __builtin_memcpy(&ua, &ha, 2);
    __builtin_memcpy(&ub, &hb, 2);
    return (uint32_t)ua | ((uint32_t)ub << 16);
}

// ---------------- weight pack kernel ----------------
// Packs W^T (K x N, with W stored [N][K] row-major fp32) into WMMA B-fragment
// layout: dst dword index = ((nt*KT + kt)*256 + lane*8 + v)
// B 32x16 layout: lane l holds col N=l%16; K0 = kt*32 + (v/4)*16 + (l/16)*8 + (v%4)*2.
__global__ void pack_b_kernel(const float* __restrict__ W, uint32_t* __restrict__ dst,
                              int Kdim, int KT, int NT) {
    const int idx = blockIdx.x * 256 + threadIdx.x;
    const int total = KT * NT * 256;
    if (idx >= total) return;
    const int v = idx & 7;
    const int l = (idx >> 3) & 31;
    const int frag = idx >> 8;
    const int kt = frag % KT;
    const int nt = frag / KT;
    const int K0 = kt * 32 + (v >> 2) * 16 + (l >> 4) * 8 + (v & 3) * 2;
    const int n  = nt * 16 + (l & 15);
    const float f0 = W[(size_t)n * Kdim + K0];
    const float f1 = W[(size_t)n * Kdim + K0 + 1];
    dst[idx] = pk2bf(f0, f1);
}

// ---------------- main fused LSTM autoencoder kernel ----------------
// grid = 16 blocks (16 batch rows each), block = 256 threads (8 wave32).
// Wave w owns H-column tiles {2w, 2w+1}; c state stays in its VGPRs.
__global__ __launch_bounds__(256) void lstm_ae_main(
    const float* __restrict__ x, const uint32_t* __restrict__ ws,
    const float* __restrict__ encbih, const float* __restrict__ encbhh,
    const float* __restrict__ attnW, const float* __restrict__ attnb,
    const float* __restrict__ fcb, const float* __restrict__ bng,
    const float* __restrict__ bnb, const float* __restrict__ bnm,
    const float* __restrict__ bnv, const float* __restrict__ dhb,
    const float* __restrict__ dcb, const float* __restrict__ decbih,
    const float* __restrict__ decbhh, const float* __restrict__ outb,
    float* __restrict__ out) {
    __shared__ __align__(16) __hip_bfloat16 sA[16 * 256];  // h / context (A operand)
    __shared__ __align__(16) __hip_bfloat16 sX[16 * 128];  // x_t / emb / rec
    __shared__ float sAcc[16 * 256];                       // online attention acc
    __shared__ float sAW[256];                             // attn_W staged
    __shared__ float sRed[16][16];
    __shared__ float sM[16], sL[16], sCoef[16], sP[16];

    const int t = threadIdx.x;
    const int lane = t & 31;
    const int wave = t >> 5;
    const int l15 = lane & 15;
    const int half = lane >> 4;
    const int b0 = blockIdx.x * 16;

    const uint32_t* sAdw = (const uint32_t*)sA;
    const uint32_t* sXdw = (const uint32_t*)sX;
    const uint32_t* pkEncWih = ws + PK_ENC_WIH;
    const uint32_t* pkEncWhh = ws + PK_ENC_WHH;
    const uint32_t* pkDecWih = ws + PK_DEC_WIH;
    const uint32_t* pkDecWhh = ws + PK_DEC_WHH;
    const uint32_t* pkFc = ws + PK_FC;
    const uint32_t* pkDh = ws + PK_DH;
    const uint32_t* pkDc = ws + PK_DC;
    const uint32_t* pkOut = ws + PK_OUT;

    float* scores = out + WGT_OFF;

    // ---- init ----
    const __hip_bfloat16 bz = __float2bfloat16(0.0f);
    for (int i = t; i < 16 * 256; i += 256) {
        sA[i] = bz;
        sAcc[i] = 0.0f;
    }
    sAW[t] = attnW[t];
    if (t < 16) {
        sM[t] = -1e30f;
        sL[t] = 0.0f;
    }
    const float attnb0 = attnb[0];

    // hoisted, loop-invariant per-thread gate biases (encoder)
    float encB[2][4];
#pragma unroll
    for (int j = 0; j < 2; ++j)
#pragma unroll
        for (int g = 0; g < 4; ++g) {
            const int n = (g * 16 + (2 * wave + j)) * 16 + l15;
            encB[j][g] = encbih[n] + encbhh[n];
        }

    v8f cfr[2];
#pragma unroll
    for (int j = 0; j < 2; ++j)
#pragma unroll
        for (int r = 0; r < 8; ++r) cfr[j][r] = 0.0f;
    __syncthreads();

    // ================= encoder =================
#pragma unroll 1
    for (int s = 0; s < SS; ++s) {
        // stage x_t as bf16 [16][64] : one float4 per thread
        {
            const int row = t >> 4, col4 = (t & 15) * 4;
            const float4 xv = *(const float4*)(x + (size_t)(b0 + row) * (SS * FF) +
                                               (size_t)s * FF + col4);
            uint32_t* sXd = (uint32_t*)sX;
            sXd[t * 2] = pk2bf(xv.x, xv.y);
            sXd[t * 2 + 1] = pk2bf(xv.z, xv.w);
        }
        __syncthreads();

        // gates = x_t @ Wih^T + h @ Whh^T + (bih + bhh)
        v8f acc[2][4];
#pragma unroll
        for (int j = 0; j < 2; ++j)
#pragma unroll
            for (int g = 0; g < 4; ++g)
#pragma unroll
                for (int r = 0; r < 8; ++r) acc[j][g][r] = encB[j][g];

#pragma unroll 1
        for (int kt = 0; kt < 2; ++kt) {  // x contribution (rolled: no spills)
            const v16bf a = load_a_frag(sXdw, 32, kt, lane);
#pragma unroll
            for (int j = 0; j < 2; ++j) {
                const int ht = 2 * wave + j;
#pragma unroll
                for (int g = 0; g < 4; ++g)
                    acc[j][g] = wmma_bf16(
                        a, load_b_frag(pkEncWih, (g * 16 + ht) * 2 + kt, lane),
                        acc[j][g]);
            }
        }
#pragma unroll 1
        for (int kt = 0; kt < 8; ++kt) {  // h contribution (rolled: no spills)
            const v16bf a = load_a_frag(sAdw, 128, kt, lane);
#pragma unroll
            for (int j = 0; j < 2; ++j) {
                const int ht = 2 * wave + j;
#pragma unroll
                for (int g = 0; g < 4; ++g)
                    acc[j][g] = wmma_bf16(
                        a, load_b_frag(pkEncWhh, (g * 16 + ht) * 8 + kt, lane),
                        acc[j][g]);
            }
        }
        __syncthreads();  // all sA/sX fragment reads done

        // LSTM cell nonlinearity; c stays in VGPRs; h -> sA (bf16)
#pragma unroll
        for (int j = 0; j < 2; ++j) {
            const int ht = 2 * wave + j;
#pragma unroll
            for (int r = 0; r < 8; ++r) {
                const float iv = fsig(acc[j][0][r]);
                const float fv = fsig(acc[j][1][r]);
                const float gv = ftanh_(acc[j][2][r]);
                const float ov = fsig(acc[j][3][r]);
                const float c = fv * cfr[j][r] + iv * gv;
                cfr[j][r] = c;
                const float h = ov * ftanh_(c);
                const int M = half * 8 + r;
                sA[M * 256 + ht * 16 + l15] = __float2bfloat16(h);
            }
        }
        __syncthreads();  // h_s ready

        // attention score: s_b = h_b . attn_w + attn_b   (online softmax)
        {
            const int b = t >> 4, seg = t & 15;
            float p = 0.0f;
#pragma unroll
            for (int k = 0; k < 16; ++k)
                p += __bfloat162float(sA[b * 256 + seg * 16 + k]) * sAW[seg * 16 + k];
            sRed[b][seg] = p;
        }
        __syncthreads();
        if (t < 16) {
            float sc = attnb0;
#pragma unroll
            for (int k = 0; k < 16; ++k) sc += sRed[t][k];
            scores[(size_t)(b0 + t) * SS + s] = sc;  // raw score, normalized later
            const float m_old = sM[t];
            const float m_new = fmaxf(m_old, sc);
            const float coef = __expf(m_old - m_new);
            const float p = __expf(sc - m_new);
            sM[t] = m_new;
            sCoef[t] = coef;
            sP[t] = p;
            sL[t] = sL[t] * coef + p;
        }
        __syncthreads();
        {
            const int b = t >> 4, c0 = (t & 15) * 16;
            const float coef = sCoef[b], p = sP[b];
#pragma unroll
            for (int k = 0; k < 16; ++k) {
                const int cc = b * 256 + c0 + k;
                sAcc[cc] = sAcc[cc] * coef + p * __bfloat162float(sA[cc]);
            }
        }
    }
    __syncthreads();

    // ---- context = acc / l  -> sA (bf16 A operand) ----
    {
        const int b = t >> 4, c0 = (t & 15) * 16;
        const float invl = __builtin_amdgcn_rcpf(sL[b]);
#pragma unroll
        for (int k = 0; k < 16; ++k)
            sA[b * 256 + c0 + k] = __float2bfloat16(sAcc[b * 256 + c0 + k] * invl);
    }
    __threadfence();
    __syncthreads();

    // ---- finalize attention weights output ----
    {
        const int b = t >> 4;
        const int s0 = (t & 15) * 32;
        const float m = sM[b];
        const float invl = __builtin_amdgcn_rcpf(sL[b]);
#pragma unroll 1
        for (int k = 0; k < 32; ++k) {
            const size_t idx = (size_t)(b0 + b) * SS + s0 + k;
            scores[idx] = __expf(scores[idx] - m) * invl;
        }
    }

    // ---- emb = tanh(bn(context @ fc_W^T + fc_b)) ; wave = E n-tile ----
    {
        const int n = wave * 16 + l15;
        const float fb = fcb[n];
        v8f ea;
#pragma unroll
        for (int r = 0; r < 8; ++r) ea[r] = fb;
#pragma unroll 1
        for (int kt = 0; kt < 8; ++kt) {
            const v16bf a = load_a_frag(sAdw, 128, kt, lane);
            ea = wmma_bf16(a, load_b_frag(pkFc, wave * 8 + kt, lane), ea);
        }
        const float scale = bng[n] * rsqrtf(bnv[n] + 1e-5f);
        const float shift = bnb[n] - bnm[n] * scale;
#pragma unroll
        for (int r = 0; r < 8; ++r) {
            const float e = ftanh_(ea[r] * scale + shift);
            const int M = half * 8 + r;
            out[EMB_OFF + (size_t)(b0 + M) * EE + n] = e;
            sX[M * 128 + n] = __float2bfloat16(e);  // emb as next A operand
        }
    }
    __syncthreads();

    // ---- decoder init: dec_h = emb@dh^T + dh_b -> sA ; dec_c -> c regs ----
    {
        v8f ha[2];
#pragma unroll
        for (int j = 0; j < 2; ++j) {
            const int n = (2 * wave + j) * 16 + l15;
            const float bh = dhb[n], bc = dcb[n];
#pragma unroll
            for (int r = 0; r < 8; ++r) {
                ha[j][r] = bh;
                cfr[j][r] = bc;
            }
        }
#pragma unroll 1
        for (int kt = 0; kt < 4; ++kt) {
            const v16bf a = load_a_frag(sXdw, 64, kt, lane);  // emb [16][128]
#pragma unroll
            for (int j = 0; j < 2; ++j) {
                const int nt = 2 * wave + j;
                ha[j] = wmma_bf16(a, load_b_frag(pkDh, nt * 4 + kt, lane), ha[j]);
                cfr[j] = wmma_bf16(a, load_b_frag(pkDc, nt * 4 + kt, lane), cfr[j]);
            }
        }
        __syncthreads();  // emb reads done before sX reuse
#pragma unroll
        for (int j = 0; j < 2; ++j) {
            const int ht = 2 * wave + j;
#pragma unroll
            for (int r = 0; r < 8; ++r) {
                const int M = half * 8 + r;
                sA[M * 256 + ht * 16 + l15] = __float2bfloat16(ha[j][r]);
            }
        }
        // inp0 = zeros [16][64]
        for (int i = t; i < 16 * 64; i += 256) sX[i] = bz;
    }

    // hoisted decoder gate biases + out bias
    float decB[2][4];
#pragma unroll
    for (int j = 0; j < 2; ++j)
#pragma unroll
        for (int g = 0; g < 4; ++g) {
            const int n = (g * 16 + (2 * wave + j)) * 16 + l15;
            decB[j][g] = decbih[n] + decbhh[n];
        }
    const float outB = outb[wave < 4 ? wave * 16 + l15 : l15];
    __syncthreads();

    // ================= decoder (autoregressive) =================
#pragma unroll 1
    for (int s = 0; s < SS; ++s) {
        v8f acc[2][4];
#pragma unroll
        for (int j = 0; j < 2; ++j)
#pragma unroll
            for (int g = 0; g < 4; ++g)
#pragma unroll
                for (int r = 0; r < 8; ++r) acc[j][g][r] = decB[j][g];

#pragma unroll 1
        for (int kt = 0; kt < 2; ++kt) {  // input (rec) contribution
            const v16bf a = load_a_frag(sXdw, 32, kt, lane);
#pragma unroll
            for (int j = 0; j < 2; ++j) {
                const int ht = 2 * wave + j;
#pragma unroll
                for (int g = 0; g < 4; ++g)
                    acc[j][g] = wmma_bf16(
                        a, load_b_frag(pkDecWih, (g * 16 + ht) * 2 + kt, lane),
                        acc[j][g]);
            }
        }
#pragma unroll 1
        for (int kt = 0; kt < 8; ++kt) {  // h contribution
            const v16bf a = load_a_frag(sAdw, 128, kt, lane);
#pragma unroll
            for (int j = 0; j < 2; ++j) {
                const int ht = 2 * wave + j;
#pragma unroll
                for (int g = 0; g < 4; ++g)
                    acc[j][g] = wmma_bf16(
                        a, load_b_frag(pkDecWhh, (g * 16 + ht) * 8 + kt, lane),
                        acc[j][g]);
            }
        }
        __syncthreads();

#pragma unroll
        for (int j = 0; j < 2; ++j) {
            const int ht = 2 * wave + j;
#pragma unroll
            for (int r = 0; r < 8; ++r) {
                const float iv = fsig(acc[j][0][r]);
                const float fv = fsig(acc[j][1][r]);
                const float gv = ftanh_(acc[j][2][r]);
                const float ov = fsig(acc[j][3][r]);
                const float c = fv * cfr[j][r] + iv * gv;
                cfr[j][r] = c;
                const float h = ov * ftanh_(c);
                const int M = half * 8 + r;
                sA[M * 256 + ht * 16 + l15] = __float2bfloat16(h);
            }
        }
        __syncthreads();  // h ready; sX free

        // rec = h @ out_W^T + out_b  (waves 0..3, one F n-tile each)
        if (wave < 4) {
            const int n = wave * 16 + l15;
            v8f ra;
#pragma unroll
            for (int r = 0; r < 8; ++r) ra[r] = outB;
#pragma unroll 1
            for (int kt = 0; kt < 8; ++kt) {
                const v16bf a = load_a_frag(sAdw, 128, kt, lane);
                ra = wmma_bf16(a, load_b_frag(pkOut, wave * 8 + kt, lane), ra);
            }
#pragma unroll
            for (int r = 0; r < 8; ++r) {
                const int M = half * 8 + r;
                const float rv = ra[r];
                out[RECON_OFF + ((size_t)(b0 + M) * SS + s) * FF + n] = rv;
                sX[M * 64 + n] = __float2bfloat16(rv);  // feedback input
            }
        }
        __syncthreads();
    }
}

// ---------------- launch ----------------
extern "C" void kernel_launch(void* const* d_in, const int* in_sizes, int n_in,
                              void* d_out, int out_size, void* d_ws, size_t ws_size,
                              hipStream_t stream) {
    (void)in_sizes; (void)n_in; (void)out_size; (void)ws_size;
    uint32_t* ws = (uint32_t*)d_ws;

    // pack weights (fp32 -> bf16 WMMA B-fragments), all tiny
    pack_b_kernel<<<128, 256, 0, stream>>>((const float*)d_in[1],  ws + PK_ENC_WIH, 64,  2, 64);
    pack_b_kernel<<<512, 256, 0, stream>>>((const float*)d_in[2],  ws + PK_ENC_WHH, 256, 8, 64);
    pack_b_kernel<<<128, 256, 0, stream>>>((const float*)d_in[17], ws + PK_DEC_WIH, 64,  2, 64);
    pack_b_kernel<<<512, 256, 0, stream>>>((const float*)d_in[18], ws + PK_DEC_WHH, 256, 8, 64);
    pack_b_kernel<<<64,  256, 0, stream>>>((const float*)d_in[7],  ws + PK_FC,      256, 8, 8);
    pack_b_kernel<<<64,  256, 0, stream>>>((const float*)d_in[13], ws + PK_DH,      128, 4, 16);
    pack_b_kernel<<<64,  256, 0, stream>>>((const float*)d_in[15], ws + PK_DC,      128, 4, 16);
    pack_b_kernel<<<32,  256, 0, stream>>>((const float*)d_in[21], ws + PK_OUT,     256, 8, 4);

    lstm_ae_main<<<16, 256, 0, stream>>>(
        (const float*)d_in[0], ws,
        (const float*)d_in[3],  (const float*)d_in[4],   // enc biases
        (const float*)d_in[5],  (const float*)d_in[6],   // attn
        (const float*)d_in[8],                            // fc_b
        (const float*)d_in[9],  (const float*)d_in[10],
        (const float*)d_in[11], (const float*)d_in[12],  // bn
        (const float*)d_in[14], (const float*)d_in[16],  // dh_b, dc_b
        (const float*)d_in[19], (const float*)d_in[20],  // dec biases
        (const float*)d_in[22],                           // out_b
        (float*)d_out);
}
